// DifferentiableSMC_71193377899255
// MI455X (gfx1250) — matrix operand order repaired and checked by reading, not compiled, not run on hardware
//
#include <hip/hip_runtime.h>
#include <cstdint>
#include <cstddef>

// ---------------------------------------------------------------------------
// Differentiable SMC forward (lml only) for MI455X / gfx1250.
//   - bf16 WMMA (v_wmma_f32_16x16x32_bf16) for the two per-step GEMMs
//   - B fragments pre-swizzled to lane order; staged to LDS via TDM
//     (tensor_load_to_lds + s_wait_tensorcnt) in the hot kernel
//   - particles stored transposed [64][N] for b128 stores / coalesced reads
// ---------------------------------------------------------------------------

#define ND    64     // STATE_DIM
#define NOBS  32     // OBS_DIM
#define NPART 32768  // NUM_PARTICLES
#define NSTEP 64     // T

static constexpr float DT_C   = 0.01f;
static constexpr float DAMP   = 0.99f;
static constexpr float INVT   = 2.0f;                 // 1/TEMPERATURE
static constexpr float LOG2PI = 1.8378770664093453f;

typedef __attribute__((ext_vector_type(16))) __bf16 v16bf;
typedef __attribute__((ext_vector_type(8)))  __bf16 v8bf;
typedef __attribute__((ext_vector_type(8)))  float  v8f;
typedef unsigned int u32x4 __attribute__((ext_vector_type(4)));
typedef int          i32x8 __attribute__((ext_vector_type(8)));
typedef int          i32x4 __attribute__((ext_vector_type(4)));

// ---------------- workspace layout (bytes) ----------------
// PT: particles transposed [64][NPART] f32
static constexpr size_t P_OFF    = 0;
static constexpr size_t LOGW_OFF = P_OFF    + (size_t)NPART*ND*4;      // [N] f32
static constexpr size_t SCAL_OFF = LOGW_OFF + (size_t)NPART*4;         // [0]=S,[1]=lml,[2]=c0
static constexpr size_t MEAN_OFF = SCAL_OFF + 256;                     // mean[64]
static constexpr size_t V_OFF    = MEAN_OFF + 256;                     // v[(T+1)*32] f32
static constexpr size_t FRAG_OFF = V_OFF    + ((size_t)(NSTEP+1)*NOBS*4 + 128);
// FRAG: pre-swizzled WMMA B fragments, bf16:
//   M frags  [jt=2][kt=2][lane=32][h=16]  -> 2048 bf16 (4 KB)
//   Qt frags [dt=4][kt=2][lane=32][h=16]  -> 4096 bf16 (8 KB)
static constexpr int    FRAG_M_V16   = 4 * 32;         // 128 v16bf entries
static constexpr int    FRAG_TOT_V16 = 12 * 32;        // 384 v16bf = 12 KB
static constexpr size_t PART_OFF = FRAG_OFF + (size_t)FRAG_TOT_V16 * 32;
static constexpr size_t LSEP_OFF = PART_OFF + (size_t)128*66*4 + 256;

// ---------------- deterministic counter RNG ----------------
__device__ __forceinline__ uint32_t ihash(uint32_t x) {
  x ^= x >> 16; x *= 0x7feb352dU; x ^= x >> 15; x *= 0x846ca68bU; x ^= x >> 16;
  return x;
}
__device__ __forceinline__ float u01(uint32_t h) {
  return (float)(h >> 8) * 0x1p-24f + 0x1p-25f;   // (0,1)
}
__device__ __forceinline__ float rnorm(uint64_t idx, uint32_t seed) {
  uint32_t lo = (uint32_t)idx, hi = (uint32_t)(idx >> 32);
  uint32_t h1 = ihash(lo ^ (hi * 0x9E3779B9u) ^ seed);
  uint32_t h2 = ihash(h1 + 0x6a09e667u);
  return __builtin_sqrtf(-2.0f * __logf(u01(h1))) * __cosf(6.2831853071795864f * u01(h2));
}
__device__ __forceinline__ void rnorm2(uint64_t idx, uint32_t seed, float& z0, float& z1) {
  uint32_t lo = (uint32_t)idx, hi = (uint32_t)(idx >> 32);
  uint32_t h1 = ihash(lo ^ (hi * 0x9E3779B9u) ^ seed);
  uint32_t h2 = ihash(h1 + 0x6a09e667u);
  const float r = __builtin_sqrtf(-2.0f * __logf(u01(h1)));
  float s, c;
  __sincosf(6.2831853071795864f * u01(h2), &s, &c);
  z0 = r * c; z1 = r * s;
}

// ---------------- WMMA helpers ----------------
// A-fragment (16x32 bf16): lane holds row m = lane&15; halves -> K:
//   h<8 : K = (lane>=16?8:0)+h ;  h>=8 : K = 16 + (lane>=16?8:0) + (h-8)
template <typename F>
__device__ __forceinline__ void load_Atiles(F getElem, v16bf& a0, v16bf& a1) {
  const int lane = threadIdx.x & 31;
  const int m  = lane & 15;
  const int g8 = (lane >> 4) * 8;
#pragma unroll
  for (int h = 0; h < 16; ++h) {
    const int kk = (h < 8) ? (g8 + h) : (16 + g8 + (h - 8));
    a0[h] = getElem(m, kk);
    a1[h] = getElem(m, 32 + kk);
  }
}

// Likelihood from A tiles against pre-swizzled M fragments (v16bf, indexed
// [jt*2+kt][lane]):  z = v - A*M^T ; logw = c0 - 0.5*|z|^2
template <typename VP>
__device__ __forceinline__ void logw_from_Atiles(
    const v16bf a0, const v16bf a1, VP fragM,
    const float* __restrict__ vvec, float c0, float* __restrict__ logw_out, int row0) {
  const int lane = threadIdx.x & 31;
  const int g    = lane >> 4;
  const int l15  = lane & 15;

  float sq[8];
#pragma unroll
  for (int r = 0; r < 8; ++r) sq[r] = 0.0f;

#pragma unroll
  for (int jt = 0; jt < 2; ++jt) {
    const v16bf b0 = fragM[(jt * 2 + 0) * 32 + lane];
    const v16bf b1 = fragM[(jt * 2 + 1) * 32 + lane];
    v8f d;
#pragma unroll
    for (int i = 0; i < 8; ++i) d[i] = 0.0f;
    d = __builtin_amdgcn_wmma_f32_16x16x32_bf16(false, a0, false, b0, (short)0, d, false, false);
    d = __builtin_amdgcn_wmma_f32_16x16x32_bf16(false, a1, false, b1, (short)0, d, false, false);
    const float vj = vvec[jt * 16 + l15];
#pragma unroll
    for (int r = 0; r < 8; ++r) { float z = vj - d[r]; sq[r] += z * z; }
  }
#pragma unroll
  for (int r = 0; r < 8; ++r) {
    float s = sq[r];
#pragma unroll
    for (int o = 8; o > 0; o >>= 1) s += __shfl_xor(s, o, 32);
    if (l15 == 0) logw_out[row0 + r + 8 * g] = c0 - 0.5f * s;
  }
}

// ---------------- kernels ----------------

// One-block setup: Cholesky of R, Linv, c0, v[t] = Linv*obs[t],
// and pre-swizzled bf16 B fragments for both GEMMs.
__global__ void k_setup(const float* __restrict__ obs, const float* __restrict__ C,
                        const float* __restrict__ Qc, const float* __restrict__ Rc,
                        float* __restrict__ scal, float* __restrict__ vout,
                        __bf16* __restrict__ frag) {
  __shared__ float Rm[NOBS * NOBS];
  __shared__ float Li[NOBS * NOBS];
  const int tid = threadIdx.x;

  for (int idx = tid; idx < NOBS * NOBS; idx += 256) {
    const int i = idx / NOBS, j = idx % NOBS;
    float s = 0.0f;
    for (int k = 0; k < NOBS; ++k) s += Rc[i * NOBS + k] * Rc[j * NOBS + k];
    Rm[idx] = s;
  }
  __syncthreads();

  if (tid == 0) {
    for (int j = 0; j < NOBS; ++j) {                  // lower Cholesky in-place
      float d = Rm[j * NOBS + j];
      for (int k = 0; k < j; ++k) { float l = Rm[j * NOBS + k]; d -= l * l; }
      d = __builtin_sqrtf(fmaxf(d, 1e-20f));
      Rm[j * NOBS + j] = d;
      for (int i = j + 1; i < NOBS; ++i) {
        float s = Rm[i * NOBS + j];
        for (int k = 0; k < j; ++k) s -= Rm[i * NOBS + k] * Rm[j * NOBS + k];
        Rm[i * NOBS + j] = s / d;
      }
    }
    float ld = 0.0f;
    for (int i = 0; i < NOBS; ++i) ld += __logf(Rm[i * NOBS + i]);
    scal[2] = -0.5f * (float)NOBS * LOG2PI - ld;      // c0
    scal[1] = 0.0f;                                   // lml accumulator
    for (int j = 0; j < NOBS; ++j) {                  // Linv (forward subst.)
      for (int i = 0; i < j; ++i) Li[i * NOBS + j] = 0.0f;
      Li[j * NOBS + j] = 1.0f / Rm[j * NOBS + j];
      for (int i = j + 1; i < NOBS; ++i) {
        float s = 0.0f;
        for (int k = j; k < i; ++k) s += Rm[i * NOBS + k] * Li[k * NOBS + j];
        Li[i * NOBS + j] = -s / Rm[i * NOBS + i];
      }
    }
  }
  __syncthreads();

  // M fragments: B[k][n]=M[j][k], j=jt*16+(lane&15), k=kt*32+table(h,lane>>4)
  for (int idx = tid; idx < 4 * 32 * 16; idx += 256) {
    const int h = idx & 15, lane = (idx >> 4) & 31, f = idx >> 9;
    const int jt = f >> 1, kt = f & 1;
    const int j = jt * 16 + (lane & 15);
    const int k = kt * 32 + ((h < 8) ? (((lane >> 4) * 8) + h) : (16 + ((lane >> 4) * 8) + (h - 8)));
    float s = 0.0f;
    for (int kk = 0; kk <= j; ++kk) s += Li[j * NOBS + kk] * C[kk * ND + k];
    frag[idx] = (__bf16)s;
  }
  // Qt fragments: B[k][d] = Qc^T[k][d] = Qc[d][k]
  for (int idx = tid; idx < 8 * 32 * 16; idx += 256) {
    const int h = idx & 15, lane = (idx >> 4) & 31, f = idx >> 9;
    const int dt = f >> 1, kt = f & 1;
    const int d = dt * 16 + (lane & 15);
    const int k = kt * 32 + ((h < 8) ? (((lane >> 4) * 8) + h) : (16 + ((lane >> 4) * 8) + (h - 8)));
    frag[2048 + idx] = (__bf16)Qc[d * ND + k];
  }
  // v[t] = Linv @ obs[t]
  for (int idx = tid; idx < (NSTEP + 1) * NOBS; idx += 256) {
    const int t = idx / NOBS, j = idx % NOBS;
    float s = 0.0f;
    for (int k = 0; k <= j; ++k) s += Li[j * NOBS + k] * obs[t * NOBS + k];
    vout[idx] = s;
  }
}

// initial particles ~ N(0,1) (transposed layout is an arbitrary iid relabel)
__global__ void k_init(float* __restrict__ PT) {
  const size_t i = (size_t)blockIdx.x * 256 + threadIdx.x;   // over 64*N
  PT[i] = rnorm(i, 0x1234567u);
}

// standalone likelihood for t = 0 (A from transposed particles in global)
__global__ void k_logw(const float* __restrict__ PT, const __bf16* __restrict__ fragG,
                       const float* __restrict__ vvec, const float* __restrict__ scal,
                       float* __restrict__ logw) {
  const int wave = blockIdx.x * (blockDim.x >> 5) + (threadIdx.x >> 5);
  const int row0 = wave * 16;
  v16bf a0, a1;
  load_Atiles([&](int m, int k) -> __bf16 {
      return (__bf16)PT[(size_t)k * NPART + row0 + m]; }, a0, a1);
  logw_from_Atiles(a0, a1, (const v16bf*)fragG, vvec, scal[2], logw, row0);
}

// fused propagate + likelihood
__global__ void k_prop(int t, const __bf16* __restrict__ fragG,
                       const float* __restrict__ mean, const float* __restrict__ vvec,
                       const float* __restrict__ scal, float* __restrict__ PT,
                       float* __restrict__ logw) {
  __shared__ __attribute__((aligned(32))) __bf16 sFrag[FRAG_TOT_V16 * 16]; // 12 KB
  __shared__ __bf16 stile[8][ND * 16];                                     // 16 KB (transposed [K][m])
  const int wid  = threadIdx.x >> 5;
  const int lane = threadIdx.x & 31;
  const int wave = blockIdx.x * 8 + wid;
  const int row0 = wave * 16;
  const int g    = lane >> 4;
  const int l15  = lane & 15;
  const int g8   = g * 8;

#if defined(__has_builtin)
#if __has_builtin(__builtin_amdgcn_tensor_load_to_lds) && __has_builtin(__builtin_amdgcn_s_wait_tensorcnt)
#define SMC_TDM 1
#endif
#endif
#ifdef SMC_TDM
  if (threadIdx.x < 32) {
    // TDM: DMA 12 KB of pre-swizzled fragments (global -> LDS), 1536 x 8B
    const uint64_t ga = (uint64_t)(uintptr_t)fragG;
    u32x4 g0;
    g0[0] = 1u;                                       // count=1 (valid D#)
    g0[1] = (uint32_t)(uintptr_t)&sFrag[0];           // lds_addr
    g0[2] = (uint32_t)ga;                             // global_addr[31:0]
    g0[3] = ((uint32_t)(ga >> 32) & 0x01FFFFFFu) | (2u << 30); // addr[56:32] | type=2
    i32x8 g1;
    g1[0] = (int)(3u << 16);                          // data_size = 8B
    g1[1] = (int)(1536u << 16);                       // tensor_dim0[15:0]
    g1[2] = (int)(1u << 16);                          // dim0 hi | tensor_dim1 = 1
    g1[3] = (int)(1536u << 16);                       // dim1 hi | tile_dim0 = 1536
    g1[4] = 1;                                        // tile_dim1 = 1, tile_dim2 = 0
    g1[5] = 1536;                                     // tensor_dim0_stride lo
    g1[6] = 0; g1[7] = 0;
    i32x4 g2; g2[0] = g2[1] = g2[2] = g2[3] = 0;
    i32x4 g3; g3[0] = g3[1] = g3[2] = g3[3] = 0;
#if defined(__clang_major__) && (__clang_major__ >= 23)
    i32x8 g4;                                         // therock clang-23: 6-arg form
    g4[0] = g4[1] = g4[2] = g4[3] = g4[4] = g4[5] = g4[6] = g4[7] = 0;
    __builtin_amdgcn_tensor_load_to_lds(g0, g1, g2, g3, g4, 0);
#else
    __builtin_amdgcn_tensor_load_to_lds(g0, g1, g2, g3, 0);  // ROCm 7.2 clang-22: 5-arg
#endif
  }
#else
  for (int i = threadIdx.x; i < FRAG_TOT_V16 * 8; i += 256)
    ((uint32_t*)sFrag)[i] = ((const uint32_t*)fragG)[i];
#endif

  // Gaussian noise in A-fragment layout (Box-Muller pairs), overlapped with TDM
  v16bf n0, n1;
  {
    const uint64_t base = ((uint64_t)t * NPART + (uint64_t)(row0 + l15)) * ND;
#pragma unroll
    for (int h = 0; h < 16; h += 2) {
      const int kk = (h < 8) ? (g8 + h) : (16 + g8 + (h - 8));
      float z0, z1, z2, z3;
      rnorm2(base + kk,      0xBEEFCAFEu, z0, z1);
      rnorm2(base + 32 + kk, 0xC0FFEE11u, z2, z3);
      n0[h] = (__bf16)z0; n0[h + 1] = (__bf16)z1;
      n1[h] = (__bf16)z2; n1[h + 1] = (__bf16)z3;
    }
  }

#ifdef SMC_TDM
  if (threadIdx.x < 32) __builtin_amdgcn_s_wait_tensorcnt(0);
#endif
  __syncthreads();

  const v16bf* fragQ = (const v16bf*)&sFrag[0] + FRAG_M_V16;
  const float  md_c0 = scal[2];

#pragma unroll
  for (int dt = 0; dt < 4; ++dt) {
    const v16bf b0 = fragQ[(dt * 2 + 0) * 32 + lane];
    const v16bf b1 = fragQ[(dt * 2 + 1) * 32 + lane];
    v8f acc;
#pragma unroll
    for (int i = 0; i < 8; ++i) acc[i] = 0.0f;
    acc = __builtin_amdgcn_wmma_f32_16x16x32_bf16(false, n0, false, b0, (short)0, acc, false, false);
    acc = __builtin_amdgcn_wmma_f32_16x16x32_bf16(false, n1, false, b1, (short)0, acc, false, false);
    const int d = dt * 16 + l15;
    const float md = mean[d];
    float4 lo4, hi4;
    v8bf  pk;
#pragma unroll
    for (int r = 0; r < 8; ++r) {
      const float val = acc[r] + md;
      pk[r] = (__bf16)val;
      if (r < 4) (&lo4.x)[r] = val; else (&hi4.x)[r - 4] = val;
    }
    float* dst = &PT[(size_t)d * NPART + row0 + 8 * g];
    *(float4*)dst       = lo4;                        // global_store_b128
    *((float4*)dst + 1) = hi4;
    *(v8bf*)&stile[wid][d * 16 + 8 * g] = pk;         // 16B LDS store
  }
  __syncthreads();

  v16bf a0, a1;
  load_Atiles([&](int m, int k) -> __bf16 { return stile[wid][k * 16 + m]; }, a0, a1);
  logw_from_Atiles(a0, a1, (const v16bf*)&sFrag[0], vvec, md_c0, logw, row0);
}

// softmax-weighted-mean pass 1: per-block (max, sumexp, 64 weighted sums)
__global__ void k_wmean_p1(int t, const float* __restrict__ logw, const float* __restrict__ PT,
                           const float* __restrict__ scal, float* __restrict__ part) {
  __shared__ float accS[65];
  __shared__ float wm[8];
  __shared__ float bmax;
  const int tid = threadIdx.x;
  const int n   = blockIdx.x * 256 + tid;

  const float S = scal[0];
  const uint32_t h = ihash((uint32_t)n ^ ihash(0xA5A50000u + (uint32_t)t));
  const float gum = -__logf(-__logf(u01(h) + 1e-10f) + 1e-10f);
  const float a   = (logw[n] - S + gum) * INVT;

  float m = a;
#pragma unroll
  for (int o = 16; o > 0; o >>= 1) m = fmaxf(m, __shfl_xor(m, o, 32));
  if ((tid & 31) == 0) wm[tid >> 5] = m;
  if (tid < 65) accS[tid] = 0.0f;
  __syncthreads();
  if (tid == 0) { float mm = wm[0]; for (int i = 1; i < 8; ++i) mm = fmaxf(mm, wm[i]); bmax = mm; }
  __syncthreads();

  const float e = __expf(a - bmax);
  float se = e;
#pragma unroll
  for (int o = 16; o > 0; o >>= 1) se += __shfl_xor(se, o, 32);
  if ((tid & 31) == 0) atomicAdd(&accS[0], se);
  for (int d = 0; d < ND; ++d) {
    float w = e * PT[(size_t)d * NPART + n];          // coalesced
#pragma unroll
    for (int o = 16; o > 0; o >>= 1) w += __shfl_xor(w, o, 32);
    if ((tid & 31) == 0) atomicAdd(&accS[1 + d], w);  // ds_add_f32
  }
  __syncthreads();
  if (tid == 0)  part[blockIdx.x * 66] = bmax;
  if (tid < 65)  part[blockIdx.x * 66 + 1 + tid] = accS[tid];
}

// pass 2 (1 block): combine partials -> r[64]; drift mean (A, Q_terms)
__global__ void k_wmean_p2(const float* __restrict__ part, const float* __restrict__ A,
                           const float* __restrict__ Qt, float* __restrict__ meanOut) {
  __shared__ float sh[65];
  __shared__ float r[ND];
  __shared__ float Mx;
  const int tid = threadIdx.x;
  if (tid == 0) { float m = part[0]; for (int b = 1; b < 128; ++b) m = fmaxf(m, part[b * 66]); Mx = m; }
  if (tid < 65) sh[tid] = 0.0f;
  __syncthreads();
  for (int idx = tid; idx < 128 * 65; idx += 256) {
    const int b = idx / 65, c = idx % 65;
    atomicAdd(&sh[c], part[b * 66 + 1 + c] * __expf(part[b * 66] - Mx));
  }
  __syncthreads();
  if (tid < ND) r[tid] = sh[1 + tid] / sh[0];
  __syncthreads();
  if (tid < ND) {
    const int i = tid;
    float q = 0.0f, ar = 0.0f;
    for (int j = 0; j < ND; ++j) {
      const float rj = r[j];
      ar += A[i * ND + j] * rj;
      const float* row = Qt + ((size_t)i * ND + j) * ND;
      float acc = 0.0f;
      for (int k = 0; k < ND; ++k) acc += row[k] * r[k];
      q += rj * acc;
    }
    meanOut[i] = r[i] + DT_C * (ar + q) * DAMP;
  }
}

// logsumexp pass 1: per-block (max, sumexp)
__global__ void k_lse_p1(const float* __restrict__ logw, float* __restrict__ lpart) {
  __shared__ float wm[8];
  __shared__ float wsum[8];
  __shared__ float bm;
  const int tid = threadIdx.x;
  const float a = logw[blockIdx.x * 256 + tid];
  float m = a;
#pragma unroll
  for (int o = 16; o > 0; o >>= 1) m = fmaxf(m, __shfl_xor(m, o, 32));
  if ((tid & 31) == 0) wm[tid >> 5] = m;
  __syncthreads();
  if (tid == 0) { float mm = wm[0]; for (int i = 1; i < 8; ++i) mm = fmaxf(mm, wm[i]); bm = mm; }
  __syncthreads();
  float e = __expf(a - bm);
#pragma unroll
  for (int o = 16; o > 0; o >>= 1) e += __shfl_xor(e, o, 32);
  if ((tid & 31) == 0) wsum[tid >> 5] = e;
  __syncthreads();
  if (tid == 0) {
    float s = 0.0f;
    for (int i = 0; i < 8; ++i) s += wsum[i];
    lpart[blockIdx.x * 2] = bm;
    lpart[blockIdx.x * 2 + 1] = s;
  }
}

// logsumexp pass 2 (1 block, 128 threads)
__global__ void k_lse_p2(const float* __restrict__ lpart, float* __restrict__ scal,
                         int add_lml, float* __restrict__ out, int write_out) {
  __shared__ float arr[128];
  __shared__ float Mx;
  const int tid = threadIdx.x;
  arr[tid] = lpart[tid * 2];
  __syncthreads();
  if (tid == 0) { float m = arr[0]; for (int i = 1; i < 128; ++i) m = fmaxf(m, arr[i]); Mx = m; }
  __syncthreads();
  arr[tid] = lpart[tid * 2 + 1] * __expf(lpart[tid * 2] - Mx);
  __syncthreads();
  if (tid == 0) {
    float s = 0.0f;
    for (int i = 0; i < 128; ++i) s += arr[i];
    const float S = Mx + __logf(s);
    scal[0] = S;
    if (add_lml) {
      const float nl = scal[1] + S;
      scal[1] = nl;
      if (write_out) out[0] = nl;
    }
  }
}

// ---------------------------------------------------------------------------
extern "C" void kernel_launch(void* const* d_in, const int* in_sizes, int n_in,
                              void* d_out, int out_size, void* d_ws, size_t ws_size,
                              hipStream_t stream) {
  (void)in_sizes; (void)n_in; (void)out_size; (void)ws_size;
  const float* obs = (const float*)d_in[0];   // [65,32]
  const float* A   = (const float*)d_in[1];   // [64,64]
  const float* C   = (const float*)d_in[2];   // [32,64]
  const float* Qt  = (const float*)d_in[3];   // [64,64,64]
  const float* Qc  = (const float*)d_in[4];   // [64,64]
  const float* Rc  = (const float*)d_in[5];   // [32,32]

  char*   w    = (char*)d_ws;
  float*  PT   = (float*)(w + P_OFF);
  float*  logw = (float*)(w + LOGW_OFF);
  float*  scal = (float*)(w + SCAL_OFF);
  float*  mean = (float*)(w + MEAN_OFF);
  float*  vv   = (float*)(w + V_OFF);
  __bf16* frag = (__bf16*)(w + FRAG_OFF);
  float*  part = (float*)(w + PART_OFF);
  float*  lprt = (float*)(w + LSEP_OFF);
  float*  out  = (float*)d_out;

  const int wmmaBlocks = (NPART / 16) / 8;   // 256 blocks of 8 waves

  k_setup<<<1, 256, 0, stream>>>(obs, C, Qc, Rc, scal, vv, frag);
  k_init<<<(NPART * ND) / 256, 256, 0, stream>>>(PT);
  k_logw<<<wmmaBlocks, 256, 0, stream>>>(PT, frag, vv, scal, logw);
  k_lse_p1<<<128, 256, 0, stream>>>(logw, lprt);
  k_lse_p2<<<1, 128, 0, stream>>>(lprt, scal, 0, out, 0);

  for (int t = 1; t <= NSTEP; ++t) {
    k_wmean_p1<<<128, 256, 0, stream>>>(t, logw, PT, scal, part);
    k_wmean_p2<<<1, 256, 0, stream>>>(part, A, Qt, mean);
    k_prop<<<wmmaBlocks, 256, 0, stream>>>(t, frag, mean, vv + (size_t)t * NOBS, scal, PT, logw);
    k_lse_p1<<<128, 256, 0, stream>>>(logw, lprt);
    k_lse_p2<<<1, 128, 0, stream>>>(lprt, scal, 1, out, (t == NSTEP) ? 1 : 0);
  }
}
